// PPGN_82721070121708
// MI455X (gfx1250) — matrix operand order repaired and checked by reading, not compiled, not run on hardware
//
#include <hip/hip_runtime.h>

typedef __attribute__((ext_vector_type(16))) __bf16 v16bf;
typedef __attribute__((ext_vector_type(2)))  __bf16 v2bf;
typedef __attribute__((ext_vector_type(8)))  float  v8f;
typedef __attribute__((ext_vector_type(4)))  unsigned int v4u;
typedef __attribute__((ext_vector_type(8)))  unsigned int v8u;
typedef unsigned short ushort_t;
typedef unsigned int   uint_t;

#define NPIX 900
#define PIXP 912
#define NCH  40
#define NW   8          // 8 wave32 per workgroup (256 threads)

// ---- dynamic-LDS byte offsets (all 16B aligned) ----
#define OFF_PPA   0         // 76800 B  ping-pong buf A ([pix][40] 80B rows / [ch][30][32] 64B rows)
#define OFF_PPB   76800     // 76800 B  ping-pong buf B
#define OFF_X2T   153600    // 76800 B  x2 transposed [ch][j][32]
#define OFF_Y     230400    // 72960 B  x1@x2*Ms [pix][40]
#define OFF_W     303360    // 4 x 3840 B weights bf16 [40][48]: wa, wb, wc1, wc2
#define OFF_MS    318720    // 912 bf16
#define OFF_M0    320544    // 912 bf16
#define OFF_M1    322368    // 912 bf16
#define OFF_FEAT  324192    // 400 f32
#define LDS_BYTES 325792

// f32 -> bf16 via native fptrunc (v_cvt_pk_bf16_f32 on CDNA5-class HW)
__device__ __forceinline__ ushort_t cvt1(float a) {
    __bf16 h = (__bf16)a;
    return __builtin_bit_cast(ushort_t, h);
}
__device__ __forceinline__ uint_t cvt2(float a, float b) {
#if __has_builtin(__builtin_amdgcn_cvt_pk_bf16_f32)
    v2bf p = __builtin_amdgcn_cvt_pk_bf16_f32(a, b);
    return __builtin_bit_cast(uint_t, p);
#else
    v2bf p;
    p[0] = (__bf16)a;
    p[1] = (__bf16)b;
    return __builtin_bit_cast(uint_t, p);
#endif
}
__device__ __forceinline__ float bf2f(ushort_t h) {
    uint_t u = ((uint_t)h) << 16;
    return __builtin_bit_cast(float, u);
}
__device__ __forceinline__ v4u ld128(const char* p) { return *(const v4u*)p; }
__device__ __forceinline__ v4u zero4() { v4u z; z[0]=0u; z[1]=0u; z[2]=0u; z[3]=0u; return z; }
__device__ __forceinline__ v16bf mkfrag(v4u lo, v4u hi) {
    v8u u;
#pragma unroll
    for (int i = 0; i < 4; ++i) { u[i] = lo[i]; u[i + 4] = hi[i]; }
    return __builtin_bit_cast(v16bf, u);
}
__device__ __forceinline__ v8f zero8() {
    v8f z;
#pragma unroll
    for (int i = 0; i < 8; ++i) z[i] = 0.f;
    return z;
}
__device__ __forceinline__ v8f wmma_bf16(v16bf a, v16bf b, v8f c) {
    return __builtin_amdgcn_wmma_f32_16x16x32_bf16(false, a, false, b, (short)0, c, false, false);
}

// ---- fragment loaders: all unconditional ds_load_b128, padding supplies zeros ----
__device__ __forceinline__ v16bf load_A64(const char* base, int row, int half) {
    const char* p = base + row * 64 + half * 16;
    return mkfrag(ld128(p), ld128(p + 32));
}
__device__ __forceinline__ v16bf load_Aw_lo(const char* base, int row, int half) {
    const char* p = base + row * 96 + half * 16;
    return mkfrag(ld128(p), ld128(p + 32));
}
__device__ __forceinline__ v16bf load_Aw_hi(const char* base, int row, int half) {
    const char* p = base + row * 96 + 64 + half * 16;
    return mkfrag(ld128(p), zero4());
}
__device__ __forceinline__ v16bf load_B64(const char* base, int col, int half) {
    const char* p = base + col * 64 + half * 16;
    return mkfrag(ld128(p), ld128(p + 32));
}
__device__ __forceinline__ v16bf load_B80_lo(const char* base, int col, int half) {
    const char* p = base + col * 80 + half * 16;
    return mkfrag(ld128(p), ld128(p + 32));
}
__device__ __forceinline__ v16bf load_B80_hi(const char* base, int col, int half) {
    const char* p = base + col * 80 + 64 + half * 32;
    v4u lo = ld128(p);
    uint_t keep = half ? 0u : ~0u;
    lo &= keep;
    return mkfrag(lo, zero4());
}

// x1/x2 = relu((W @ x) * Ms); dst is [ch][i or j][32] (64B rows)
__device__ __forceinline__ void conv_gemm(const char* wlds, const char* src, char* dst,
                                          bool transposed, const ushort_t* MsL,
                                          int lane, int wave) {
    const int half = (lane >> 4) & 1, r = lane & 15;
    ushort_t* d16 = (ushort_t*)dst;
    for (int mt = 0; mt < 3; ++mt) {
        v16bf a0 = load_Aw_lo(wlds, mt * 16 + r, half);
        v16bf a1 = load_Aw_hi(wlds, mt * 16 + r, half);
        for (int nt = wave; nt < 57; nt += NW) {
            const int col = nt * 16 + r;
            v16bf b0 = load_B80_lo(src, col, half);
            v16bf b1 = load_B80_hi(src, col, half);
            v8f acc = zero8();
            acc = wmma_bf16(a0, b0, acc);
            acc = wmma_bf16(a1, b1, acc);
            if (col < NPIX && !(mt == 2 && half)) {
                const float ms = bf2f(MsL[col]);
                const int q = transposed ? ((col % 30) * 32 + col / 30)
                                         : ((col / 30) * 32 + col % 30);
                ushort_t* dp = d16 + (mt * 16 + 8 * half) * 960 + q;
#pragma unroll
                for (int e = 0; e < 4; ++e) {
                    uint_t pk = cvt2(fmaxf(acc[2 * e] * ms, 0.f),
                                     fmaxf(acc[2 * e + 1] * ms, 0.f));
                    dp[0]   = (ushort_t)pk;
                    dp[960] = (ushort_t)(pk >> 16);
                    dp += 1920;
                }
            }
        }
    }
}

__global__ void __launch_bounds__(256)
ppgn_fused(const float* __restrict__ X2, const float* __restrict__ M,
           const float* __restrict__ w1_1, const float* __restrict__ w1_2,
           const float* __restrict__ w1_3, const float* __restrict__ wA,
           const float* __restrict__ wB, const float* __restrict__ wC,
           const float* __restrict__ h1w, const float* __restrict__ h1b,
           float* __restrict__ out) {
    extern __shared__ __align__(16) char smem[];
    char* bufA = smem + OFF_PPA;
    char* bufB = smem + OFF_PPB;
    char* x2t  = smem + OFF_X2T;
    char* ybuf = smem + OFF_Y;
    ushort_t* wLa  = (ushort_t*)(smem + OFF_W);
    ushort_t* wLb  = wLa + 40 * 48;
    ushort_t* wLc1 = wLb + 40 * 48;
    ushort_t* wLc2 = wLc1 + 40 * 48;
    ushort_t* MsL  = (ushort_t*)(smem + OFF_MS);
    ushort_t* M0L  = (ushort_t*)(smem + OFF_M0);
    ushort_t* M1L  = (ushort_t*)(smem + OFF_M1);
    float*    feat = (float*)(smem + OFF_FEAT);

    const int b    = blockIdx.x;
    const int tid  = threadIdx.x;
    const int lane = tid & 31;
    const int wave = tid >> 5;
    const int half = (lane >> 4) & 1, r = lane & 15;

    // ---- init: masks (bf16, exact for {0,1,2}), x0, x2t k-padding ----
    for (int p = tid; p < PIXP; p += 256) {
        float m0 = (p < NPIX) ? M[(size_t)(b * 2 + 0) * NPIX + p] : 0.f;
        float m1 = (p < NPIX) ? M[(size_t)(b * 2 + 1) * NPIX + p] : 0.f;
        M0L[p] = cvt1(m0); M1L[p] = cvt1(m1); MsL[p] = cvt1(m0 + m1);
    }
    {
        ushort_t* x0 = (ushort_t*)bufA;
        for (int i = tid; i < PIXP * NCH; i += 256) {
            const int p = i / NCH, ch = i % NCH;
            float v = (ch < 3 && p < NPIX) ? X2[(size_t)(b * 3 + ch) * NPIX + p] : 0.f;
            x0[p * NCH + ch] = cvt1(v);
        }
        for (int i = tid; i < 1200; i += 256) {          // x2t pad k=30,31 (persistent)
            const int c = i / 30, row = i % 30;
            *(uint_t*)(x2t + c * 1920 + row * 64 + 60) = 0u;
        }
    }
    __syncthreads();

    char* pX  = bufA;   // current x, [pix][40] (80B rows)
    char* pX1 = bufB;   // x1 [ch][i][32] view, later xn [pix][40] view

    for (int blk = 0; blk < 5; ++blk) {
        const float *wa, *wb, *wc;
        int Cact, Kc;
        if (blk == 0) { wa = w1_1; wb = w1_2; wc = w1_3; Cact = 3;  Kc = 43; }
        else { wa = wA + (blk - 1) * 1600; wb = wB + (blk - 1) * 1600;
               wc = wC + (blk - 1) * 3200; Cact = 40; Kc = 80; }

        // ---- stage weights -> LDS bf16 [40][48] (zero-padded), zero pX1 k-pads ----
        for (int i = tid; i < 40 * 48; i += 256) {
            const int rr = i / 48, k = i % 48;
            wLa[i]  = cvt1(k < Cact ? wa[rr * Cact + k] : 0.f);
            wLb[i]  = cvt1(k < Cact ? wb[rr * Cact + k] : 0.f);
            wLc1[i] = cvt1(k < 40 ? wc[rr * Kc + k] : 0.f);
            wLc2[i] = cvt1(k < Kc - 40 ? wc[rr * Kc + 40 + k] : 0.f);
        }
        for (int i = tid; i < 1200; i += 256) {
            const int c = i / 30, row = i % 30;
            *(uint_t*)(pX1 + c * 1920 + row * 64 + 60) = 0u;
        }
        __syncthreads();

        // ---- GEMM1/2: x1 -> pX1 [ch][i][32], x2^T -> x2t [ch][j][32] ----
        conv_gemm((const char*)wLa, pX, pX1, false, MsL, lane, wave);
        conv_gemm((const char*)wLb, pX, x2t, true,  MsL, lane, wave);
        __syncthreads();

        // ---- GEMM3: per-channel (30x30)@(30x30) * Ms -> Y [pix][40] ----
        for (int job = wave; job < 160; job += NW) {
            const int c = job >> 2, q = job & 3, mtt = q >> 1, ntl = q & 1;
            v16bf a  = load_A64(pX1 + c * 1920, mtt * 16 + r, half);
            v16bf bb = load_B64(x2t + c * 1920, ntl * 16 + r, half);
            v8f acc = zero8();
            acc = wmma_bf16(a, bb, acc);
            const int j = ntl * 16 + r;
            if (j < 30) {
                ushort_t* y16 = (ushort_t*)ybuf;
                const int i0 = mtt * 16 + 8 * half;
                int p = i0 * 30 + j;
#pragma unroll
                for (int e = 0; e < 8; ++e) {
                    if (i0 + e < 30) {
                        y16[p * NCH + c] = cvt1(acc[e] * bf2f(MsL[p]));
                    }
                    p += 30;
                }
            }
        }
        __syncthreads();

        // ---- GEMM4: xn = relu((wc1@Y + wc2@x) * Ms) -> pX1 as [pix][40] ----
        for (int mt = 0; mt < 3; ++mt) {
            v16bf a0 = load_Aw_lo((const char*)wLc1, mt * 16 + r, half);
            v16bf a1 = load_Aw_hi((const char*)wLc1, mt * 16 + r, half);
            v16bf a2 = load_Aw_lo((const char*)wLc2, mt * 16 + r, half);
            v16bf a3 = load_Aw_hi((const char*)wLc2, mt * 16 + r, half);
            for (int nt = wave; nt < 57; nt += NW) {
                const int col = nt * 16 + r;
                v16bf b0 = load_B80_lo(ybuf, col, half);
                v16bf b1 = load_B80_hi(ybuf, col, half);
                v16bf b2 = load_B80_lo(pX, col, half);
                v16bf b3 = load_B80_hi(pX, col, half);
                v8f acc = zero8();
                acc = wmma_bf16(a0, b0, acc);
                acc = wmma_bf16(a1, b1, acc);
                acc = wmma_bf16(a2, b2, acc);
                acc = wmma_bf16(a3, b3, acc);
                if (col < NPIX && !(mt == 2 && half)) {
                    const float ms = bf2f(MsL[col]);
                    v4u d;
#pragma unroll
                    for (int e = 0; e < 4; ++e)
                        d[e] = cvt2(fmaxf(acc[2 * e] * ms, 0.f),
                                    fmaxf(acc[2 * e + 1] * ms, 0.f));
                    *(v4u*)(pX1 + col * 80 + (mt * 16 + 8 * half) * 2) = d;
                }
            }
        }
        __syncthreads();

        // ---- masked spatial sums -> feat[blk*80 + msel*40 + ch] ----
        for (int job = wave; job < 80; job += NW) {
            const int msel = job / NCH, ch = job % NCH;
            const ushort_t* Mm = msel ? M1L : M0L;
            const ushort_t* xn = (const ushort_t*)pX1;
            float s = 0.f;
            for (int p = lane; p < NPIX; p += 32)
                s += bf2f(xn[p * NCH + ch]) * bf2f(Mm[p]);
#pragma unroll
            for (int off = 16; off >= 1; off >>= 1)
                s += __shfl_xor(s, off);
            if (lane == 0) feat[blk * 80 + job] = s;
        }
        __syncthreads();

        char* t = pX; pX = pX1; pX1 = t;   // xn becomes next x
    }

    // ---- final: out = relu(feat @ h1_w^T + h1_b), 400 -> 32 ----
    if (tid < 32) {
        float s = h1b[tid];
        for (int f = 0; f < 400; ++f)
            s += feat[f] * h1w[tid * 400 + f];
        out[(size_t)b * 32 + tid] = s > 0.f ? s : 0.f;
    }
}

extern "C" void kernel_launch(void* const* d_in, const int* in_sizes, int n_in,
                              void* d_out, int out_size, void* d_ws, size_t ws_size,
                              hipStream_t stream) {
    const float* X2   = (const float*)d_in[0];
    const float* M    = (const float*)d_in[1];
    const float* w1_1 = (const float*)d_in[2];
    const float* w1_2 = (const float*)d_in[3];
    const float* w1_3 = (const float*)d_in[4];
    const float* wA   = (const float*)d_in[5];
    const float* wB   = (const float*)d_in[6];
    const float* wC   = (const float*)d_in[7];
    const float* h1w  = (const float*)d_in[8];
    const float* h1b  = (const float*)d_in[9];
    float* out = (float*)d_out;

    const int B = in_sizes[0] / (3 * NPIX);
    (void)hipFuncSetAttribute((const void*)ppgn_fused,
                              hipFuncAttributeMaxDynamicSharedMemorySize, LDS_BYTES);
    ppgn_fused<<<B, 256, LDS_BYTES, stream>>>(X2, M, w1_1, w1_2, w1_3, wA, wB, wC,
                                              h1w, h1b, out);
}
